// NonLocalBlock_31379031064663
// MI455X (gfx1250) — matrix operand order repaired and checked by reading, not compiled, not run on hardware
//
#include <hip/hip_runtime.h>
#include <stdint.h>

// NonLocalBlock for MI455X (gfx1250, wave32, WMMA).
// B=8 L=2048 D=512 H=512 E=256 K=30
#define B_ 8
#define L_ 2048
#define D_ 512
#define H_ 512
#define E_ 256
#define K_ 30
#define M_ (B_*L_)      // 16384 rows total
#define NCAT 1024       // H + E + E fused projection width

typedef __bf16 bf16_t;
typedef __attribute__((ext_vector_type(16))) __bf16 v16bf;
typedef __attribute__((ext_vector_type(8)))  __bf16 v8bf;
typedef __attribute__((ext_vector_type(8)))  float  v8f;

__device__ __forceinline__ bf16_t f2bf(float f) {
  uint32_t u = __builtin_bit_cast(uint32_t, f);
  uint32_t r = u + 0x7FFFu + ((u >> 16) & 1u);   // round-to-nearest-even
  uint16_t b = (uint16_t)(r >> 16);
  return __builtin_bit_cast(bf16_t, b);
}

#define WMMA_BF16(a, b, c) \
  __builtin_amdgcn_wmma_f32_16x16x32_bf16(false, (a), false, (b), (short)0, (c), false, false)

// A-fragment loader per CDNA5 ISA 16-bit A 16x32 layout:
// lanes 0-15 hold M=0-15, K = {0..7, 16..23}; lanes 16-31 hold K = {8..15, 24..31}.
__device__ __forceinline__ v16bf load_a16(const bf16_t* __restrict__ rowptr, int kbase, int lane) {
  int ko = kbase + ((lane >> 4) << 3);
  v8bf lo = *(const v8bf*)(rowptr + ko);
  v8bf hi = *(const v8bf*)(rowptr + ko + 16);
  return __builtin_shufflevector(lo, hi, 0,1,2,3,4,5,6,7,8,9,10,11,12,13,14,15);
}

__device__ __forceinline__ v16bf load_b16(const bf16_t* __restrict__ colptr, int ld, int lane) {
  int ko = (lane >> 4) << 3;
  v16bf r;
  for (int s = 0; s < 8; ++s) {
    r[s] = colptr[(size_t)(ko + s) * ld];
    r[s + 8] = colptr[(size_t)(ko + 16 + s) * ld];
  }
  return r;
}

// ---------------- pack / convert kernels ----------------
__global__ void pack_x(const float* __restrict__ x, bf16_t* __restrict__ Xb, int n) {
  int i = blockIdx.x * blockDim.x + threadIdx.x;
  if (i < n) Xb[i] = f2bf(x[i]);
}

__global__ void pack_w(const float* __restrict__ wg,  const float* __restrict__ wgb,
                       const float* __restrict__ t1,  const float* __restrict__ t1b,
                       const float* __restrict__ t2,  const float* __restrict__ t2b,
                       const float* __restrict__ wz,
                       bf16_t* __restrict__ Wcat, float* __restrict__ bcat,
                       bf16_t* __restrict__ Wzb) {
  int i = blockIdx.x * blockDim.x + threadIdx.x;
  if (i < D_ * NCAT) {                  // fused [D, H|E|E] weight
    int k = i / NCAT, n = i % NCAT;
    float v;
    if (n < H_)            v = wg[k * H_ + n];
    else if (n < H_ + E_)  v = t1[k * E_ + (n - H_)];
    else                   v = t2[k * E_ + (n - H_ - E_)];
    Wcat[i] = f2bf(v);
  }
  int j = i - D_ * NCAT;
  if (j >= 0 && j < NCAT) {
    bcat[j] = (j < H_) ? wgb[j] : (j < H_ + E_ ? t1b[j - H_] : t2b[j - H_ - E_]);
  }
  int m = j - NCAT;
  if (m >= 0 && m < H_ * D_) Wzb[m] = f2bf(wz[m]);
}

// ---------------- fused input projections: X @ [Wg|T1|T2] ----------------
// One wave per 16x64 output tile: 4 independent accumulator chains, each A
// fragment reused 4x (0.5 b128 A loads per WMMA instead of 2; B uses 16 strided 2-byte loads).
// x_prime stored f32 (gather source), x1 bf16 row-major (attention A side),
// x2 stored TRANSPOSED [B,E,L] so attention B loads are contiguous across lanes (stride L within a lane).
__global__ void __launch_bounds__(256) gemm_in(
    const bf16_t* __restrict__ Xb,    // [M, D]
    const bf16_t* __restrict__ Wcat,  // [D, NCAT]
    const float*  __restrict__ bcat,  // [NCAT]
    float*        __restrict__ x_prime, // [M, H]
    bf16_t*       __restrict__ x1b,     // [M, E]
    bf16_t*       __restrict__ x2T)     // [B, E, L]
{
  int lane = threadIdx.x & 31;
  int wid  = (blockIdx.x * blockDim.x + threadIdx.x) >> 5;
  const int NB = NCAT / 64;             // 16 column blocks of 64
  int mt = wid / NB, nb = wid % NB;
  int row0 = mt * 16, n0 = nb * 64;
  const bf16_t* arow = Xb + (size_t)(row0 + (lane & 15)) * D_;
  v8f acc[4] = {v8f{}, v8f{}, v8f{}, v8f{}};
  for (int k0 = 0; k0 < D_; k0 += 32) {
    const bf16_t* bp = Wcat + (size_t)k0 * NCAT + n0 + (lane & 15);  // lane%16 = N
    if (k0 + 32 < D_) {
      __builtin_prefetch(bp + 32 * NCAT, 0, 1);
      __builtin_prefetch(bp + 32 * NCAT + 48, 0, 1);
    }
    v16bf a = load_a16(arow, k0, lane);
    v16bf bf0 = load_b16(bp, NCAT, lane);
    v16bf bf1 = load_b16(bp + 16, NCAT, lane);
    v16bf bf2 = load_b16(bp + 32, NCAT, lane);
    v16bf bf3 = load_b16(bp + 48, NCAT, lane);
    acc[0] = WMMA_BF16(a, bf0, acc[0]);
    acc[1] = WMMA_BF16(a, bf1, acc[1]);
    acc[2] = WMMA_BF16(a, bf2, acc[2]);
    acc[3] = WMMA_BF16(a, bf3, acc[3]);
  }
  int mbase = row0 + ((lane >> 4) << 3);  // C/D layout: lanes>=16 -> M+8
  int nlo = lane & 15;
  // 64-column block lies entirely inside one output segment (all multiples of 64)
  if (n0 < H_) {
    for (int c = 0; c < 4; ++c) {
      int n = n0 + c * 16 + nlo;
      float bias = bcat[n];
      for (int r = 0; r < 8; ++r)
        x_prime[(size_t)(mbase + r) * H_ + n] = acc[c][r] + bias;
    }
  } else if (n0 < H_ + E_) {
    for (int c = 0; c < 4; ++c) {
      int n = n0 + c * 16 + nlo;
      float bias = bcat[n];
      int e = n - H_;
      for (int r = 0; r < 8; ++r)
        x1b[(size_t)(mbase + r) * E_ + e] = f2bf(acc[c][r] + bias);
    }
  } else {
    int bb = mbase / L_, lp0 = mbase % L_;
    for (int c = 0; c < 4; ++c) {
      int n = n0 + c * 16 + nlo;
      float bias = bcat[n];
      int e = n - (H_ + E_);
      bf16_t* dst = x2T + ((size_t)bb * E_ + e) * L_ + lp0;
      for (int r = 0; r < 8; ++r)
        dst[r] = f2bf(acc[c][r] + bias);
    }
  }
}

// ---------------- fused attention: logits + mask + top-30 softmax + gather ----
// One block (4 waves) per 16-row tile. Logit row block [16 x 2048] f32 lives in
// LDS (128 KB of the WGP's 320 KB). softmax->mask->renorm->topk->renorm
// algebraically reduces to: softmax over the top-30 masked logits.
__global__ void __launch_bounds__(128) attn_topk(
    const bf16_t* __restrict__ x1b,     // [M, E]
    const bf16_t* __restrict__ x2T,     // [B, E, L]
    const float*  __restrict__ x_prime, // [M, H]
    const int*    __restrict__ x_len,   // [B]
    bf16_t*       __restrict__ yb)      // [M, H]
{
  __shared__ float zsh[16 * L_];        // 131072 bytes
  __shared__ int   topidx[16 * K_];
  __shared__ float topw[16 * K_];

  int lane = threadIdx.x & 31;
  int wv   = threadIdx.x >> 5;
  int row0 = blockIdx.x * 16;
  int b    = row0 / L_;
  int len  = x_len[b];

  // preload all A fragments for this tile's 16 query rows (K = E = 256)
  const bf16_t* arow = x1b + (size_t)(row0 + (lane & 15)) * E_;
  v16bf afr[E_ / 32];
  for (int k8 = 0; k8 < E_ / 32; ++k8) afr[k8] = load_a16(arow, k8 * 32, lane);

  const bf16_t* x2base = x2T + (size_t)b * E_ * L_;
  // j-tiles of 64 columns: 4 independent accumulators, A fragments reused 4x
  for (int jt = wv; jt < L_ / 64; jt += 4) {
    int j0 = jt * 64;
    v8f acc[4] = {v8f{}, v8f{}, v8f{}, v8f{}};
    for (int k8 = 0; k8 < E_ / 32; ++k8) {
      const bf16_t* bp = x2base + (size_t)(k8 * 32) * L_ + j0 + (lane & 15);
      __builtin_prefetch(bp + 32 * L_, 0, 1);
      v16bf bf0 = load_b16(bp, L_, lane);
      v16bf bf1 = load_b16(bp + 16, L_, lane);
      v16bf bf2 = load_b16(bp + 32, L_, lane);
      v16bf bf3 = load_b16(bp + 48, L_, lane);
      acc[0] = WMMA_BF16(afr[k8], bf0, acc[0]);
      acc[1] = WMMA_BF16(afr[k8], bf1, acc[1]);
      acc[2] = WMMA_BF16(afr[k8], bf2, acc[2]);
      acc[3] = WMMA_BF16(afr[k8], bf3, acc[3]);
    }
    int rbase = (lane >> 4) << 3;
    for (int c = 0; c < 4; ++c) {
      int j = j0 + c * 16 + (lane & 15);
      bool ok = (j < len);
      for (int r = 0; r < 8; ++r)
        zsh[(rbase + r) * L_ + j] = ok ? acc[c][r] : -1e30f;   // key-side mask
    }
  }
  __syncthreads();

  // wave wv owns rows 4*wv .. 4*wv+3
  for (int rr = 0; rr < 4; ++rr) {
    int r = wv * 4 + rr;
    float* zr = zsh + r * L_;
    const int span = L_ / 32;           // 64 candidates per lane
    int base = lane * span;
    float lmax = -3e30f; int lidx = base;
    for (int i = 0; i < span; ++i) { float v = zr[base + i]; if (v > lmax) { lmax = v; lidx = base + i; } }

    float zmx = 0.f, wsum = 0.f;
    for (int k = 0; k < K_; ++k) {
      float v = lmax; int id = lidx;
      for (int off = 16; off > 0; off >>= 1) {          // wave32 argmax butterfly
        float ov = __shfl_xor(v, off, 32);
        int   oi = __shfl_xor(id, off, 32);
        if (ov > v || (ov == v && oi < id)) { v = ov; id = oi; }
      }
      if (k == 0) zmx = v;
      float w = __expf(v - zmx);
      wsum += w;
      if (lane == 0) { topidx[r * K_ + k] = id; topw[r * K_ + k] = w; }
      if (id >= base && id < base + span) {             // owner removes & rescans
        zr[id] = -3e30f;
        lmax = -3e30f; lidx = base;
        for (int i = 0; i < span; ++i) { float v2 = zr[base + i]; if (v2 > lmax) { lmax = v2; lidx = base + i; } }
      }
    }

    // y[row] = sum_k (w_k / wsum) * x_prime[b, idx_k, :]   (sparse: 30 rows)
    float accv[H_ / 32];
    for (int h = 0; h < H_ / 32; ++h) accv[h] = 0.f;
    float inv = 1.0f / wsum;
    for (int k = 0; k < K_; ++k) {
      int id  = topidx[r * K_ + k];
      float w = topw[r * K_ + k] * inv;
      const float* xp = x_prime + ((size_t)b * L_ + id) * H_;
      for (int h = 0; h < H_ / 32; ++h) accv[h] += w * xp[lane + h * 32];
    }
    bf16_t* yrow = yb + (size_t)(row0 + r) * H_;
    for (int h = 0; h < H_ / 32; ++h) yrow[lane + h * 32] = f2bf(accv[h]);
  }
}

// ---------------- output projection + residual: out = x + Y @ Wz + b ----------
// One wave per 16x64 tile, same widened structure as gemm_in.
__global__ void __launch_bounds__(256) gemm_out(
    const bf16_t* __restrict__ Yb,    // [M, H]
    const bf16_t* __restrict__ Wz,    // [H, D]
    const float*  __restrict__ wz_b,  // [D]
    const float*  __restrict__ x_in,  // [M, D]
    float*        __restrict__ out)   // [M, D]
{
  int lane = threadIdx.x & 31;
  int wid  = (blockIdx.x * blockDim.x + threadIdx.x) >> 5;
  const int NB = D_ / 64;             // 8 column blocks
  int mt = wid / NB, nb = wid % NB;
  int row0 = mt * 16, n0 = nb * 64;
  const bf16_t* arow = Yb + (size_t)(row0 + (lane & 15)) * H_;
  v8f acc[4] = {v8f{}, v8f{}, v8f{}, v8f{}};
  for (int k0 = 0; k0 < H_; k0 += 32) {
    const bf16_t* bp = Wz + (size_t)k0 * D_ + n0 + (lane & 15);
    if (k0 + 32 < H_) {
      __builtin_prefetch(bp + 32 * D_, 0, 1);
      __builtin_prefetch(bp + 32 * D_ + 48, 0, 1);
    }
    v16bf a = load_a16(arow, k0, lane);
    v16bf bf0 = load_b16(bp, D_, lane);
    v16bf bf1 = load_b16(bp + 16, D_, lane);
    v16bf bf2 = load_b16(bp + 32, D_, lane);
    v16bf bf3 = load_b16(bp + 48, D_, lane);
    acc[0] = WMMA_BF16(a, bf0, acc[0]);
    acc[1] = WMMA_BF16(a, bf1, acc[1]);
    acc[2] = WMMA_BF16(a, bf2, acc[2]);
    acc[3] = WMMA_BF16(a, bf3, acc[3]);
  }
  int mbase = row0 + ((lane >> 4) << 3);
  int nlo = lane & 15;
  for (int c = 0; c < 4; ++c) {
    int n = n0 + c * 16 + nlo;
    float bias = wz_b[n];
    for (int r = 0; r < 8; ++r) {
      int m = mbase + r;
      out[(size_t)m * D_ + n] = acc[c][r] + bias + x_in[(size_t)m * D_ + n];
    }
  }
}

extern "C" void kernel_launch(void* const* d_in, const int* in_sizes, int n_in,
                              void* d_out, int out_size, void* d_ws, size_t ws_size,
                              hipStream_t stream) {
  const float* x    = (const float*)d_in[0];
  const float* wg   = (const float*)d_in[1];
  const float* wgb  = (const float*)d_in[2];
  const float* t1   = (const float*)d_in[3];
  const float* t1b  = (const float*)d_in[4];
  const float* t2   = (const float*)d_in[5];
  const float* t2b  = (const float*)d_in[6];
  const float* wz   = (const float*)d_in[7];
  const float* wzb  = (const float*)d_in[8];
  const int*   xlen = (const int*)d_in[9];
  float* out = (float*)d_out;

  char* ws = (char*)d_ws;
  size_t off = 0;
  auto wsalloc = [&](size_t bytes) -> void* {
    void* p = ws + off;
    off += (bytes + 255) & ~(size_t)255;
    return p;
  };
  bf16_t* Xb   = (bf16_t*)wsalloc((size_t)M_ * D_ * 2);       // 16.8 MB
  bf16_t* Wcat = (bf16_t*)wsalloc((size_t)D_ * NCAT * 2);     //  1.0 MB
  float*  bcat = (float*) wsalloc((size_t)NCAT * 4);
  bf16_t* Wzb  = (bf16_t*)wsalloc((size_t)H_ * D_ * 2);       //  0.5 MB
  float*  xp   = (float*) wsalloc((size_t)M_ * H_ * 4);       // 33.6 MB
  bf16_t* x1b  = (bf16_t*)wsalloc((size_t)M_ * E_ * 2);       //  8.4 MB
  bf16_t* x2T  = (bf16_t*)wsalloc((size_t)B_ * E_ * L_ * 2);  //  8.4 MB
  bf16_t* Yb   = (bf16_t*)wsalloc((size_t)M_ * H_ * 2);       // 16.8 MB
  (void)ws_size; (void)in_sizes; (void)n_in; (void)out_size;  // ~85.5 MB total

  int nx = M_ * D_;
  pack_x<<<(nx + 255) / 256, 256, 0, stream>>>(x, Xb, nx);
  int nw = D_ * NCAT + NCAT + H_ * D_;
  pack_w<<<(nw + 255) / 256, 256, 0, stream>>>(wg, wgb, t1, t1b, t2, t2b, wz, Wcat, bcat, Wzb);

  // 1024 M-tiles x 16 N-blocks(64 wide), 8 waves per 256-thread block
  gemm_in<<<(M_ / 16) * (NCAT / 64) / 8, 256, 0, stream>>>(Xb, Wcat, bcat, xp, x1b, x2T);
  // one block per 16-row tile
  attn_topk<<<M_ / 16, 128, 0, stream>>>(x1b, x2T, xp, xlen, Yb);
  // 1024 M-tiles x 8 N-blocks(64 wide)
  gemm_out<<<(M_ / 16) * (D_ / 64) / 8, 256, 0, stream>>>(Yb, Wzb, wzb, x, out);
}
